// VQVAE_5927054868997
// MI455X (gfx1250) — matrix operand order repaired and checked by reading, not compiled, not run on hardware
//
#include <hip/hip_runtime.h>
#include <hip/hip_bf16.h>

// ---------------- CDNA5 WMMA types & helpers ----------------
typedef __bf16 bf16_t;
typedef bf16_t v16bf __attribute__((ext_vector_type(16)));
typedef float  v8f   __attribute__((ext_vector_type(8)));

union BFrag {
    v16bf          v;
    unsigned short u[16];
    unsigned int   p[8];
};

__device__ __forceinline__ unsigned int f2bf(float f) {
    union { float f; unsigned int u; } c;
    c.f = f;
    unsigned int u = c.u;
    u += 0x7FFFu + ((u >> 16) & 1u);   // round-to-nearest-even
    return u >> 16;
}

// Tile geometry: block tile M=64 x N=128, K-step 32.
// 8 waves (2 wrow x 4 wcol); each wave owns 2 M-subtiles x 2 N-subtiles
// -> 4 wmma per k-step on 8 ds_load_b128 fragment loads.
#define MT   64
#define NT   128
#define KS   32
#define LDK  40   // row stride in u16: 80B -> fragment chunks 16B aligned

__device__ __forceinline__ void frag_from_lds(BFrag& f,
                                              const unsigned short* c0,
                                              const unsigned short* c1) {
    *(uint4*)&f.p[0] = *(const uint4*)c0;   // ds_load_b128
    *(uint4*)&f.p[4] = *(const uint4*)c1;   // ds_load_b128
}

#define WMMA_BF16(acc, A, B) \
    acc = __builtin_amdgcn_wmma_f32_16x16x32_bf16( \
              false, (A).v, false, (B).v, (short)0, acc, false, false)

// ============================================================
// Forward conv as implicit GEMM on v_wmma_f32_16x16x32_bf16
//   M = Cout, N = output pixels (n,ho,wo), K = Cin*KH*KW
// ============================================================
template<int KH, int KW, int S, int P>
__global__ __launch_bounds__(256) void conv_fwd_wmma(
    const float* __restrict__ X, const float* __restrict__ Wt,
    const float* __restrict__ Bs, float* __restrict__ Y,
    int Cin, int Hin, int Win, int Cout, int Hout, int Wout,
    int Ktot, int relu)
{
    __shared__ unsigned short lA[MT][LDK];   // [m][k]  fragment-ready
    __shared__ unsigned short lB[NT][LDK];   // [n][k]  fragment-ready (transposed)
    __shared__ int sXo[NT];                  // n * Cin*Hin*Win
    __shared__ int sHB[NT], sWB[NT];         // ho*S-P, wo*S-P
    __shared__ int sYB[NT];                  // n*Cout*HW + ho*Wout + wo

    constexpr int KHW = KH * KW;
    const int t    = threadIdx.x;
    const int lane = t & 31;
    const int wid  = t >> 5;
    const int wrow = wid >> 2;          // 0..1
    const int wcol = wid & 3;           // 0..3
    const int mBase   = blockIdx.y * MT;
    const int pixBase = blockIdx.x * NT;
    const int HW  = Hout * Wout;
    const int CHW = Hin * Win;

    // ---- per-column decode, once ----
    if (t < NT) {
        int pix = pixBase + t;
        int n   = pix / HW;
        int pr  = pix - n * HW;
        int ho  = pr / Wout, wo = pr - ho * Wout;
        sXo[t] = n * Cin * CHW;
        sHB[t] = ho * S - P;
        sWB[t] = wo * S - P;
        sYB[t] = n * Cout * HW + ho * Wout + wo;
    }
    __syncthreads();

    v8f acc00 = {}, acc01 = {}, acc10 = {}, acc11 = {};
    const int ksteps = (Ktot + KS - 1) / KS;

    for (int ks = 0; ks < ksteps; ++ks) {
        const int k0 = ks * KS;
        // ---- stage A: 64x32, pairs -> ds_store_b32 (1024 pairs) ----
        #pragma unroll
        for (int i = 0; i < 4; ++i) {
            int pidx = i * 256 + t;
            int ml = pidx >> 4, kp = (pidx & 15) << 1;
            int m = mBase + ml, kk = k0 + kp;
            int o0 = (kk     < Ktot) ? m * Ktot + kk     : 0;
            int o1 = (kk + 1 < Ktot) ? m * Ktot + kk + 1 : 0;
            unsigned int lo = (kk     < Ktot) ? f2bf(Wt[o0]) : 0u;
            unsigned int hi = (kk + 1 < Ktot) ? f2bf(Wt[o1]) : 0u;
            *(unsigned int*)&lA[ml][kp] = lo | (hi << 16);
        }
        // ---- stage B: 32x128 im2col, transposed, pairs along k (2048 pairs) ----
        #pragma unroll
        for (int i = 0; i < 8; ++i) {
            int pidx = i * 256 + t;
            int nl = pidx >> 4, kp = (pidx & 15) << 1;
            unsigned int pk = 0;
            #pragma unroll
            for (int e = 0; e < 2; ++e) {
                int kk = k0 + kp + e;
                int ci  = kk / KHW;
                int rem = kk - ci * KHW;
                int kh  = rem / KW, kw = rem - kh * KW;
                int hi  = sHB[nl] + kh;
                int wi  = sWB[nl] + kw;
                bool ok = (kk < Ktot) & ((unsigned)hi < (unsigned)Hin)
                                      & ((unsigned)wi < (unsigned)Win);
                int off = ok ? (sXo[nl] + (ci * Hin + hi) * Win + wi) : 0;
                float v = X[off];
                pk |= (ok ? f2bf(v) : 0u) << (16 * e);
            }
            *(unsigned int*)&lB[nl][kp] = pk;
        }
        __syncthreads();

        // ---- fragments: 2x ds_load_b128 each ----
        const int mll = lane & 15;
        const int hiL = lane >> 4;
        BFrag a0, a1, b0, b1;
        frag_from_lds(a0, &lA[wrow * 16 + mll][hiL * 8],
                          &lA[wrow * 16 + mll][16 + hiL * 8]);
        frag_from_lds(a1, &lA[32 + wrow * 16 + mll][hiL * 8],
                          &lA[32 + wrow * 16 + mll][16 + hiL * 8]);
        const int nc0 = wcol * 16 + mll;
        frag_from_lds(b0, &lB[nc0][hiL * 8], &lB[nc0][16 + hiL * 8]);
        frag_from_lds(b1, &lB[64 + nc0][hiL * 8], &lB[64 + nc0][16 + hiL * 8]);

        WMMA_BF16(acc00, a0, b0);
        WMMA_BF16(acc01, a0, b1);
        WMMA_BF16(acc10, a1, b0);
        WMMA_BF16(acc11, a1, b1);
        __syncthreads();
    }

    // ---- epilogue: bias + optional ReLU ----
    const int nc0 = wcol * 16 + (lane & 15);
    const int hiL = lane >> 4;
    const int yb0 = sYB[nc0];
    const int yb1 = sYB[64 + nc0];
    #pragma unroll
    for (int r = 0; r < 8; ++r) {
        int m0 = mBase + wrow * 16 + hiL * 8 + r;
        int m1 = m0 + 32;
        float b0v = Bs[m0], b1v = Bs[m1];
        float v00 = acc00[r] + b0v, v01 = acc01[r] + b0v;
        float v10 = acc10[r] + b1v, v11 = acc11[r] + b1v;
        if (relu) {
            v00 = v00 > 0.f ? v00 : 0.f;  v01 = v01 > 0.f ? v01 : 0.f;
            v10 = v10 > 0.f ? v10 : 0.f;  v11 = v11 > 0.f ? v11 : 0.f;
        }
        Y[yb0 + m0 * HW] = v00;  Y[yb1 + m0 * HW] = v01;
        Y[yb0 + m1 * HW] = v10;  Y[yb1 + m1 * HW] = v11;
    }
}

// ============================================================
// ConvTranspose2d (k=4,s=2,p=1) gather form, 4 parity classes
// (blockIdx.z). K = Cin*4, kh=kh0+2*th, kw=kw0+2*tw.
// ============================================================
__global__ __launch_bounds__(256) void deconv_s2k4_wmma(
    const float* __restrict__ X, const float* __restrict__ Wt,
    const float* __restrict__ Bs, float* __restrict__ Y,
    int Cin, int Hin, int Win, int Cout, int Hout, int Wout, int relu)
{
    __shared__ unsigned short lA[MT][LDK];
    __shared__ unsigned short lB[NT][LDK];
    __shared__ int sXo[NT];                  // n * Cin*Hin*Win
    __shared__ int sHB[NT], sWB[NT];         // ho+1, wo+1
    __shared__ int sYB[NT];

    const int pho = blockIdx.z >> 1, pwo = blockIdx.z & 1;
    const int kh0 = 1 - pho, kw0 = 1 - pwo;
    const int Ho2 = Hout >> 1, Wo2 = Wout >> 1;
    const int HW2 = Ho2 * Wo2;
    const int HW  = Hout * Wout;
    const int Ktot = Cin * 4;                // multiple of 32

    const int t    = threadIdx.x;
    const int lane = t & 31;
    const int wid  = t >> 5;
    const int wrow = wid >> 2;
    const int wcol = wid & 3;
    const int mBase   = blockIdx.y * MT;
    const int pixBase = blockIdx.x * NT;

    if (t < NT) {
        int pix = pixBase + t;
        int n   = pix / HW2;
        int pr  = pix - n * HW2;
        int ho  = (pr / Wo2) * 2 + pho;
        int wo  = (pr % Wo2) * 2 + pwo;
        sXo[t] = n * Cin * Hin * Win;
        sHB[t] = ho + 1;
        sWB[t] = wo + 1;
        sYB[t] = n * Cout * HW + ho * Wout + wo;
    }
    __syncthreads();

    v8f acc00 = {}, acc01 = {}, acc10 = {}, acc11 = {};
    const int ksteps = Ktot / KS;

    for (int ks = 0; ks < ksteps; ++ks) {
        const int k0 = ks * KS;
        // ---- A: w[ci, m, kh, kw] gather, paired ----
        #pragma unroll
        for (int i = 0; i < 4; ++i) {
            int pidx = i * 256 + t;
            int ml = pidx >> 4, kp = (pidx & 15) << 1;
            int m = mBase + ml;
            unsigned int pk = 0;
            #pragma unroll
            for (int e = 0; e < 2; ++e) {
                int kk = k0 + kp + e;
                int ci = kk >> 2, tt = kk & 3;
                int kh = kh0 + ((tt >> 1) << 1);
                int kw = kw0 + ((tt & 1) << 1);
                pk |= f2bf(Wt[((ci * Cout + m) * 4 + kh) * 4 + kw]) << (16 * e);
            }
            *(unsigned int*)&lA[ml][kp] = pk;
        }
        // ---- B: dilated-input taps, transposed, paired ----
        #pragma unroll
        for (int i = 0; i < 8; ++i) {
            int pidx = i * 256 + t;
            int nl = pidx >> 4, kp = (pidx & 15) << 1;
            unsigned int pk = 0;
            #pragma unroll
            for (int e = 0; e < 2; ++e) {
                int kk = k0 + kp + e;
                int ci = kk >> 2, tt = kk & 3;
                int kh = kh0 + ((tt >> 1) << 1);
                int kw = kw0 + ((tt & 1) << 1);
                int nh = sHB[nl] - kh, nw = sWB[nl] - kw;   // always even
                int hi = nh >> 1, wi = nw >> 1;
                bool ok = (nh >= 0) & (nw >= 0) & (hi < Hin) & (wi < Win);
                int off = ok ? (sXo[nl] + (ci * Hin + hi) * Win + wi) : 0;
                float v = X[off];
                pk |= (ok ? f2bf(v) : 0u) << (16 * e);
            }
            *(unsigned int*)&lB[nl][kp] = pk;
        }
        __syncthreads();

        const int mll = lane & 15;
        const int hiL = lane >> 4;
        BFrag a0, a1, b0, b1;
        frag_from_lds(a0, &lA[wrow * 16 + mll][hiL * 8],
                          &lA[wrow * 16 + mll][16 + hiL * 8]);
        frag_from_lds(a1, &lA[32 + wrow * 16 + mll][hiL * 8],
                          &lA[32 + wrow * 16 + mll][16 + hiL * 8]);
        const int nc0 = wcol * 16 + mll;
        frag_from_lds(b0, &lB[nc0][hiL * 8], &lB[nc0][16 + hiL * 8]);
        frag_from_lds(b1, &lB[64 + nc0][hiL * 8], &lB[64 + nc0][16 + hiL * 8]);

        WMMA_BF16(acc00, a0, b0);
        WMMA_BF16(acc01, a0, b1);
        WMMA_BF16(acc10, a1, b0);
        WMMA_BF16(acc11, a1, b1);
        __syncthreads();
    }

    const int nc0 = wcol * 16 + (lane & 15);
    const int hiL = lane >> 4;
    const int yb0 = sYB[nc0];
    const int yb1 = sYB[64 + nc0];
    #pragma unroll
    for (int r = 0; r < 8; ++r) {
        int m0 = mBase + wrow * 16 + hiL * 8 + r;
        int m1 = m0 + 32;
        float b0v = Bs[m0], b1v = Bs[m1];
        float v00 = acc00[r] + b0v, v01 = acc01[r] + b0v;
        float v10 = acc10[r] + b1v, v11 = acc11[r] + b1v;
        if (relu) {
            v00 = v00 > 0.f ? v00 : 0.f;  v01 = v01 > 0.f ? v01 : 0.f;
            v10 = v10 > 0.f ? v10 : 0.f;  v11 = v11 > 0.f ? v11 : 0.f;
        }
        Y[yb0 + m0 * HW] = v00;  Y[yb1 + m0 * HW] = v01;
        Y[yb0 + m1 * HW] = v10;  Y[yb1 + m1 * HW] = v11;
    }
}

// ============================================================
// VQ: ||e_k||^2 precompute
// ============================================================
__global__ void vq_norm(const float* __restrict__ emb, float* __restrict__ nE,
                        int K, int D)
{
    int k = blockIdx.x * blockDim.x + threadIdx.x;
    if (k < K) {
        float s = 0.0f;
        for (int d = 0; d < D; ++d) { float e = emb[k * D + d]; s += e * e; }
        nE[k] = s;
    }
}

// ============================================================
// VQ assign: WMMA z.E^T (K=64 -> 2 k-steps), argmin(||e||^2-2dot)
// with wave32 butterfly (first-index tiebreak). emb bf16 in LDS.
// Block = 256 = 8 waves; each wave owns one 16-row tile.
// ============================================================
__global__ __launch_bounds__(256) void vq_assign_wmma(
    const float* __restrict__ Z, const float* __restrict__ emb,
    const float* __restrict__ nE, float* __restrict__ idx_out,
    float* __restrict__ zq_out)
{
    __shared__ unsigned short lE[512 * 64];   // [code][d] bf16, row = 128B

    const int t = threadIdx.x;
    for (int i = t; i < 512 * 32; i += 256) {   // paired staging
        float e0 = emb[2 * i], e1 = emb[2 * i + 1];
        *(unsigned int*)&lE[2 * i] = f2bf(e0) | (f2bf(e1) << 16);
    }
    __syncthreads();

    const int lane = t & 31;
    const int wid  = t >> 5;
    const int rowTile = blockIdx.x * 8 + wid;
    const int m   = rowTile * 16 + (lane & 15);
    const int hiL = lane >> 4;

    // A fragments (rows of flat z_e), both k-steps, loaded once
    BFrag a0, a1;
    #pragma unroll
    for (int vv = 0; vv < 8; ++vv) {
        int kb = (vv < 4) ? (hiL * 8 + 2 * vv) : (16 + hiL * 8 + 2 * (vv - 4));
        a0.p[vv] = f2bf(Z[m * 64 + kb])      | (f2bf(Z[m * 64 + kb + 1])      << 16);
        a1.p[vv] = f2bf(Z[m * 64 + 32 + kb]) | (f2bf(Z[m * 64 + 32 + kb + 1]) << 16);
    }

    float bestv[8];
    int   besti[8];
    #pragma unroll
    for (int r = 0; r < 8; ++r) { bestv[r] = 3.4e38f; besti[r] = 0x7fffffff; }

    for (int ct = 0; ct < 32; ++ct) {            // 512 codes / 16
        const int code = ct * 16 + (lane & 15);
        const unsigned short* e0 = &lE[code * 64 + hiL * 8];
        BFrag b0, b1;
        frag_from_lds(b0, e0,      e0 + 16);     // d = hiL*8 .. +7, 16+hiL*8 .. +7
        frag_from_lds(b1, e0 + 32, e0 + 48);     // d = 32+hiL*8 .. +7, 48+hiL*8 .. +7
        v8f acc = {};
        WMMA_BF16(acc, a0, b0);
        WMMA_BF16(acc, a1, b1);
        const float ne = nE[code];
        #pragma unroll
        for (int r = 0; r < 8; ++r) {
            float score = ne - 2.0f * acc[r];
            if (score < bestv[r]) { bestv[r] = score; besti[r] = code; }
        }
    }

    // butterfly argmin across the 16 code-columns (all lanes converge)
    #pragma unroll
    for (int r = 0; r < 8; ++r) {
        #pragma unroll
        for (int off = 8; off >= 1; off >>= 1) {
            float ov = __shfl_xor(bestv[r], off, 16);
            int   oi = __shfl_xor(besti[r], off, 16);
            if (ov < bestv[r] || (ov == bestv[r] && oi < besti[r])) {
                bestv[r] = ov; besti[r] = oi;
            }
        }
    }

    // indices (as float) + exact f32 codeword gather into z_q
    #pragma unroll
    for (int r = 0; r < 8; ++r) {
        int bi  = besti[r];
        int row = rowTile * 16 + hiL * 8 + r;
        if ((lane & 15) == 0) idx_out[row] = (float)bi;
        #pragma unroll
        for (int j = 0; j < 4; ++j) {
            int d = (lane & 15) * 4 + j;
            zq_out[row * 64 + d] = emb[bi * 64 + d];
        }
    }
}

// ============================================================
// ConvTranspose2d (128->3, k=3, s=1, p=1) + sigmoid, tiled VALU
// (memory-bound tail; C_out=3 too small for WMMA tiles)
// ============================================================
__global__ __launch_bounds__(256) void deconv3_sigmoid(
    const float* __restrict__ X, const float* __restrict__ Wt,
    const float* __restrict__ Bs, float* __restrict__ Y,
    int Cin, int H, int Wd, int Cout)
{
    __shared__ float tile[18][18];
    const int tx = threadIdx.x & 15, ty = threadIdx.x >> 4;
    const int wo = blockIdx.x * 16 + tx;
    const int ho = blockIdx.y * 16 + ty;
    const int n  = blockIdx.z;
    const int h0 = blockIdx.y * 16 - 1, w0 = blockIdx.x * 16 - 1;

    float a0 = Bs[0], a1 = Bs[1], a2 = Bs[2];

    for (int ci = 0; ci < Cin; ++ci) {
        const float* Xc = X + (n * Cin + ci) * H * Wd;
        for (int i = threadIdx.x; i < 18 * 18; i += 256) {
            int r = i / 18, c = i - r * 18;
            int hh = h0 + r, ww = w0 + c;
            tile[r][c] = (hh >= 0 && hh < H && ww >= 0 && ww < Wd)
                             ? Xc[hh * Wd + ww] : 0.0f;
        }
        __syncthreads();
        #pragma unroll
        for (int kh = 0; kh < 3; ++kh) {
            #pragma unroll
            for (int kw = 0; kw < 3; ++kw) {
                float xv = tile[ty + 2 - kh][tx + 2 - kw];  // x[ho+1-kh, wo+1-kw]
                a0 += xv * Wt[((ci * Cout + 0) * 3 + kh) * 3 + kw];
                a1 += xv * Wt[((ci * Cout + 1) * 3 + kh) * 3 + kw];
                a2 += xv * Wt[((ci * Cout + 2) * 3 + kh) * 3 + kw];
            }
        }
        __syncthreads();
    }
    int chw  = H * Wd;
    int base = n * Cout * chw + ho * Wd + wo;
    Y[base]           = 1.0f / (1.0f + __expf(-a0));
    Y[base + chw]     = 1.0f / (1.0f + __expf(-a1));
    Y[base + 2 * chw] = 1.0f / (1.0f + __expf(-a2));
}

// ============================================================
// Orchestration
// ============================================================
extern "C" void kernel_launch(void* const* d_in, const int* in_sizes, int n_in,
                              void* d_out, int out_size, void* d_ws, size_t ws_size,
                              hipStream_t stream)
{
    const float* x   = (const float*)d_in[0];
    const float* c1w = (const float*)d_in[1];
    const float* c1b = (const float*)d_in[2];
    const float* c2w = (const float*)d_in[3];
    const float* c2b = (const float*)d_in[4];
    const float* c3w = (const float*)d_in[5];
    const float* c3b = (const float*)d_in[6];
    const float* emb = (const float*)d_in[7];
    const float* d1w = (const float*)d_in[8];
    const float* d1b = (const float*)d_in[9];
    const float* d2w = (const float*)d_in[10];
    const float* d2b = (const float*)d_in[11];
    const float* d3w = (const float*)d_in[12];
    const float* d3b = (const float*)d_in[13];

    // outputs (tuple order): x_recon | indices | z_q
    float* out     = (float*)d_out;
    float* xrec    = out;                        // 16*3*256*256   = 3,145,728
    float* idx_out = out + 3145728;              // 65,536
    float* zq_out  = out + 3145728 + 65536;      // 16*64*64*64    = 4,194,304

    // workspace layout (floats)
    float* ws = (float*)d_ws;
    float* g2 = ws;                              // 16*128*256*256 = 134,217,728
    float* h1 = ws + 134217728;                  // 16*128*128*128 =  33,554,432 (reused as g1)
    float* h2 = h1 + 33554432;                   // 16*128*64*64   =   8,388,608
    float* ze = h2 + 8388608;                    // 16*64*64*64    =   4,194,304
    float* nE = ze + 4194304;                    // 512

    // encoder
    {   // conv1: (16,3,256,256) -> (16,128,128,128), relu
        dim3 g((16 * 128 * 128) / NT, 128 / MT);
        conv_fwd_wmma<4, 4, 2, 1><<<g, 256, 0, stream>>>(x, c1w, c1b, h1,
            3, 256, 256, 128, 128, 128, 48, 1);
    }
    {   // conv2: -> (16,128,64,64), relu
        dim3 g((16 * 64 * 64) / NT, 128 / MT);
        conv_fwd_wmma<4, 4, 2, 1><<<g, 256, 0, stream>>>(h1, c2w, c2b, h2,
            128, 128, 128, 128, 64, 64, 2048, 1);
    }
    {   // conv3: -> z_e (16,64,64,64), no relu
        dim3 g((16 * 64 * 64) / NT, 64 / MT);
        conv_fwd_wmma<3, 3, 1, 1><<<g, 256, 0, stream>>>(h2, c3w, c3b, ze,
            128, 64, 64, 64, 64, 64, 1152, 0);
    }

    // vector quantizer
    vq_norm<<<2, 256, 0, stream>>>(emb, nE, 512, 64);
    vq_assign_wmma<<<4096 / 8, 256, 0, stream>>>(ze, emb, nE, idx_out, zq_out);

    // decoder
    {   // deconv1: z_q (16,64,64,64) -> g1 (16,128,128,128), relu
        dim3 g((16 * 64 * 64) / NT, 128 / MT, 4);
        deconv_s2k4_wmma<<<g, 256, 0, stream>>>(zq_out, d1w, d1b, h1,
            64, 64, 64, 128, 128, 128, 1);
    }
    {   // deconv2: -> g2 (16,128,256,256), relu
        dim3 g((16 * 128 * 128) / NT, 128 / MT, 4);
        deconv_s2k4_wmma<<<g, 256, 0, stream>>>(h1, d2w, d2b, g2,
            128, 128, 128, 128, 256, 256, 1);
    }
    {   // deconv3 + sigmoid -> x_recon
        dim3 g(256 / 16, 256 / 16, 16);
        deconv3_sigmoid<<<g, 256, 0, stream>>>(g2, d3w, d3b, xrec,
            128, 256, 256, 3);
    }
}
